// BaseEntropyCoder_68040871903265
// MI455X (gfx1250) — matrix operand order repaired and checked by reading, not compiled, not run on hardware
//
#include <hip/hip_runtime.h>
#include <hip/hip_bf16.h>

typedef _Float16 v16h __attribute__((ext_vector_type(16)));
typedef _Float16 v8h  __attribute__((ext_vector_type(8)));
typedef float    v8f  __attribute__((ext_vector_type(8)));
typedef int      v4i_vs __attribute__((vector_size(16)));   // builtin's b128 payload type

static constexpr int NB   = 8;
static constexpr int NN   = 65536;
static constexpr long NT  = (long)NB * NN;   // 524288 nodes
static constexpr int D    = 128;
static constexpr int OUTD = 256;
static constexpr int MT   = 64;              // nodes per workgroup

#if defined(__HIP_DEVICE_COMPILE__) && __has_builtin(__builtin_amdgcn_global_load_async_to_lds_b128)
#define ASYNC_LDS_OK 1
#else
#define ASYNC_LDS_OK 0
#endif

// ---------------------------------------------------------------------------
// Copy 16 bytes global -> LDS.  On toolchains exposing the gfx1250 async
// builtin this becomes GLOBAL_LOAD_ASYNC_TO_LDS_B128 (ASYNCcnt path, no VGPR
// round-trip); otherwise a plain load/ds_store pair.
// ---------------------------------------------------------------------------
__device__ inline void copy16_g2l(const _Float16* g, _Float16* l) {
#if ASYNC_LDS_OK
    auto gp = (__attribute__((address_space(1))) v4i_vs*)(unsigned long long)(uintptr_t)g;
    auto lp = (__attribute__((address_space(3))) v4i_vs*)(unsigned int)(uintptr_t)l;
    __builtin_amdgcn_global_load_async_to_lds_b128(gp, lp, 0, 0);
#else
    *(v8h*)l = *(const v8h*)g;
#endif
}

__device__ inline void wait_async_copies() {
#if ASYNC_LDS_OK
    asm volatile("s_wait_asynccnt 0x0" ::: "memory");
#endif
}

// ---------------------------------------------------------------------------
// Fragment loaders
// ---------------------------------------------------------------------------

// A fragment (16x32 f16) from an LDS tile with row stride 128 halves.
// 16-bit A layout: lanes 0-15 hold K=0..7 (v0..3) and K=16..23 (v4..7);
// lanes 16-31 hold K=8..15 and K=24..31.  Row M = lane & 15.
__device__ inline v16h lds_a_frag(const _Float16* tile, int rowBase, int kk, int lane) {
    int row = rowBase + (lane & 15);
    int lg  = lane >> 4;
    const _Float16* p = tile + row * 128 + kk * 32 + lg * 8;
    v8h lo = *(const v8h*)p;          // K = kk*32 + lg*8 .. +7
    v8h hi = *(const v8h*)(p + 16);   // K = kk*32 + 16 + lg*8 .. +7
    return __builtin_shufflevector(lo, hi, 0,1,2,3,4,5,6,7,8,9,10,11,12,13,14,15);
}

// B fragment (32x16 f16) from packed weights: fragment = 512 contiguous halves,
// lane l holds halves l*16..l*16+15 (two global_load_b128).
__device__ inline v16h ld_b_frag(const _Float16* wp, int frag, int lane) {
    return *(const v16h*)(wp + (size_t)frag * 512 + lane * 16);
}

// ---------------------------------------------------------------------------
// Weight repack: W (K x Nout, f32 row-major) -> B-fragment blocks of f16.
// One block per fragment (kk, nn); element e: lane=e>>4, half=e&15 ->
// W[kk*32 + (half<8 ? (lane>>4)*8 + half : 16 + (lane>>4)*8 + half-8)][nn*16 + (lane&15)]
// ---------------------------------------------------------------------------
__global__ __launch_bounds__(256) void pack_b_kernel(const float* __restrict__ W,
                                                     _Float16* __restrict__ out,
                                                     int Nout) {
    int nTiles = Nout >> 4;
    int kk = blockIdx.x / nTiles;
    int nn = blockIdx.x % nTiles;
    size_t base = (size_t)blockIdx.x * 512;
    #pragma unroll
    for (int i = 0; i < 2; ++i) {
        int e = threadIdx.x + i * 256;
        int l = e >> 4;
        int h = e & 15;
        int krow = kk * 32 + (h < 8 ? ((l >> 4) << 3) + h : 16 + ((l >> 4) << 3) + (h - 8));
        int ncol = (nn << 4) + (l & 15);
        out[base + e] = (_Float16)W[(size_t)krow * Nout + ncol];
    }
}

// ---------------------------------------------------------------------------
// Layer 1: f1 = relu(x @ W1_0 + b1_0) @ W1_1 + b1_1      (x = data[...,0:6])
// ---------------------------------------------------------------------------
__global__ __launch_bounds__(256) void layer1_kernel(const float* __restrict__ data,
                                                     const float* __restrict__ W1_0,
                                                     const float* __restrict__ b1_0,
                                                     const _Float16* __restrict__ W11p,
                                                     const float* __restrict__ b1_1,
                                                     _Float16* __restrict__ fout) {
    __shared__ __align__(16) float    xTile[MT * 8];
    __shared__ __align__(16) float    w0s[6 * 128];
    __shared__ __align__(16) float    b0s[128];
    __shared__ __align__(16) _Float16 hTile[MT * 128];

    const int tid  = threadIdx.x;
    const int lane = tid & 31;
    const int w    = tid >> 5;
    const long gbase = (long)blockIdx.x * MT;

    // stage inputs
    xTile[tid]       = data[gbase * 8 + tid];
    xTile[tid + 256] = data[gbase * 8 + tid + 256];
    #pragma unroll
    for (int i = 0; i < 3; ++i) w0s[tid + i * 256] = W1_0[tid + i * 256];
    if (tid < 128) b0s[tid] = b1_0[tid];
    __syncthreads();

    // h = relu(x @ W1_0 + b1_0)  -- tiny K=6, plain VALU
    #pragma unroll 4
    for (int i = 0; i < 32; ++i) {
        int idx = tid + i * 256;
        int row = idx >> 7;
        int c   = idx & 127;
        float s = b0s[c];
        #pragma unroll
        for (int j = 0; j < 6; ++j) s = fmaf(xTile[row * 8 + j], w0s[j * 128 + c], s);
        hTile[row * 128 + c] = (_Float16)fmaxf(s, 0.0f);
    }
    __syncthreads();

    // f1 = h @ W1_1 + b1_1   (K=128 -> 4 k-steps, wave w owns column tile w)
    v8f acc[4] = {};
    for (int kk = 0; kk < 4; ++kk) {
        v16h bf = ld_b_frag(W11p, kk * 8 + w, lane);
        #pragma unroll
        for (int m = 0; m < 4; ++m) {
            v16h af = lds_a_frag(hTile, m * 16, kk, lane);
            acc[m] = __builtin_amdgcn_wmma_f32_16x16x32_f16(false, af, false, bf,
                                                            (short)0, acc[m], false, false);
        }
    }
    __syncthreads();   // all waves done reading hTile

    const int col  = (w << 4) + (lane & 15);
    const int rOff = (lane >> 4) * 8;
    const float bb = b1_1[col];
    #pragma unroll
    for (int m = 0; m < 4; ++m)
        #pragma unroll
        for (int r = 0; r < 8; ++r) {
            int row = m * 16 + rOff + r;
            hTile[row * 128 + col] = (_Float16)(acc[m][r] + bb);
        }
    __syncthreads();

    // coalesced writeback
    #pragma unroll
    for (int it = 0; it < 4; ++it) {
        int idx = tid + it * 256;
        int r = idx >> 4, c = idx & 15;
        *(v8h*)(fout + (gbase + r) * 128 + c * 8) = *(const v8h*)(hTile + r * 128 + c * 8);
    }
}

// ---------------------------------------------------------------------------
// Residual block:
//   h  = relu([f, gather_parent(f)] @ W0 + b0)
//   f' = relu(h @ W1 + b1 + f)
// ---------------------------------------------------------------------------
__global__ __launch_bounds__(256) void res_block_kernel(const float* __restrict__ data,
                                                        const _Float16* __restrict__ fin,
                                                        _Float16* __restrict__ fout,
                                                        const _Float16* __restrict__ W0p,
                                                        const float* __restrict__ b0,
                                                        const _Float16* __restrict__ W1p,
                                                        const float* __restrict__ b1) {
    __shared__ __align__(16) _Float16 fTile[MT * 128];   // f (kept for residual)
    __shared__ __align__(16) _Float16 gTile[MT * 128];   // parents, then reused as h

    const int tid  = threadIdx.x;
    const int lane = tid & 31;
    const int w    = tid >> 5;
    const long gbase = (long)blockIdx.x * MT;

    // f tile: direct global -> LDS (async DMA path when available)
    #pragma unroll
    for (int it = 0; it < 4; ++it) {
        int idx = tid + it * 256;
        int r = idx >> 4, c = idx & 15;
        copy16_g2l(fin + (gbase + r) * 128 + c * 8, fTile + r * 128 + c * 8);
    }
    // parent gather (random rows), root rows -> 0 (needs predication: sync path)
    #pragma unroll
    for (int it = 0; it < 4; ++it) {
        int idx = tid + it * 256;
        int r = idx >> 4, c = idx & 15;
        long g = gbase + r;
        float pf   = data[g * 8 + 6];
        float root = data[g * 8 + 7];
        v8h v = {};
        if (root != 1.0f) {
            long b  = g >> 16;                      // N = 65536
            long pg = (b << 16) + (long)(int)pf;
            v = *(const v8h*)(fin + pg * 128 + c * 8);
        }
        *(v8h*)(gTile + r * 128 + c * 8) = v;
    }
    wait_async_copies();
    __syncthreads();

    // GEMM1: [f | f_par] (64x256) @ W0 (256x128); wave w owns column tile w
    v8f acc[4] = {};
    for (int kk = 0; kk < 8; ++kk) {
        const _Float16* src = (kk < 4) ? fTile : gTile;
        int kloc = kk & 3;
        v16h bf = ld_b_frag(W0p, kk * 8 + w, lane);
        #pragma unroll
        for (int m = 0; m < 4; ++m) {
            v16h af = lds_a_frag(src, m * 16, kloc, lane);
            acc[m] = __builtin_amdgcn_wmma_f32_16x16x32_f16(false, af, false, bf,
                                                            (short)0, acc[m], false, false);
        }
    }
    __syncthreads();   // done reading gTile (parents)

    // bias + relu -> h tile (reuse gTile)
    const int col  = (w << 4) + (lane & 15);
    const int rOff = (lane >> 4) * 8;
    {
        const float bb = b0[col];
        #pragma unroll
        for (int m = 0; m < 4; ++m)
            #pragma unroll
            for (int r = 0; r < 8; ++r) {
                int row = m * 16 + rOff + r;
                gTile[row * 128 + col] = (_Float16)fmaxf(acc[m][r] + bb, 0.0f);
            }
    }
    __syncthreads();

    // GEMM2: h (64x128) @ W1 (128x128)
    v8f acc2[4] = {};
    for (int kk = 0; kk < 4; ++kk) {
        v16h bf = ld_b_frag(W1p, kk * 8 + w, lane);
        #pragma unroll
        for (int m = 0; m < 4; ++m) {
            v16h af = lds_a_frag(gTile, m * 16, kk, lane);
            acc2[m] = __builtin_amdgcn_wmma_f32_16x16x32_f16(false, af, false, bf,
                                                             (short)0, acc2[m], false, false);
        }
    }

    // epilogue: + b1 + residual(fTile), relu; fTile cols are wave-private here
    {
        const float bb = b1[col];
        #pragma unroll
        for (int m = 0; m < 4; ++m)
            #pragma unroll
            for (int r = 0; r < 8; ++r) {
                int row = m * 16 + rOff + r;
                float v = acc2[m][r] + bb + (float)fTile[row * 128 + col];
                fTile[row * 128 + col] = (_Float16)fmaxf(v, 0.0f);
            }
    }
    __syncthreads();

    // coalesced writeback
    #pragma unroll
    for (int it = 0; it < 4; ++it) {
        int idx = tid + it * 256;
        int r = idx >> 4, c = idx & 15;
        *(v8h*)(fout + (gbase + r) * 128 + c * 8) = *(const v8h*)(fTile + r * 128 + c * 8);
    }
}

// ---------------------------------------------------------------------------
// FC: out = f4 @ Wfc (128x256) + bfc, f32 output
// ---------------------------------------------------------------------------
__global__ __launch_bounds__(256) void fc_kernel(const _Float16* __restrict__ fin,
                                                 const _Float16* __restrict__ Wfcp,
                                                 const float* __restrict__ bfc,
                                                 float* __restrict__ out) {
    __shared__ __align__(16) _Float16 fTile[MT * 128];

    const int tid  = threadIdx.x;
    const int lane = tid & 31;
    const int w    = tid >> 5;
    const long gbase = (long)blockIdx.x * MT;

    #pragma unroll
    for (int it = 0; it < 4; ++it) {
        int idx = tid + it * 256;
        int r = idx >> 4, c = idx & 15;
        copy16_g2l(fin + (gbase + r) * 128 + c * 8, fTile + r * 128 + c * 8);
    }
    wait_async_copies();
    __syncthreads();

    // wave w owns column tiles w and w+8 (16 tiles of 16 over OUT=256)
    v8f acc[2][4] = {};
    for (int kk = 0; kk < 4; ++kk) {
        v16h bf0 = ld_b_frag(Wfcp, kk * 16 + w,     lane);
        v16h bf1 = ld_b_frag(Wfcp, kk * 16 + w + 8, lane);
        #pragma unroll
        for (int m = 0; m < 4; ++m) {
            v16h af = lds_a_frag(fTile, m * 16, kk, lane);
            acc[0][m] = __builtin_amdgcn_wmma_f32_16x16x32_f16(false, af, false, bf0,
                                                               (short)0, acc[0][m], false, false);
            acc[1][m] = __builtin_amdgcn_wmma_f32_16x16x32_f16(false, af, false, bf1,
                                                               (short)0, acc[1][m], false, false);
        }
    }

    const int rOff = (lane >> 4) * 8;
    #pragma unroll
    for (int t = 0; t < 2; ++t) {
        int col = ((w + t * 8) << 4) + (lane & 15);
        float bb = bfc[col];
        #pragma unroll
        for (int m = 0; m < 4; ++m)
            #pragma unroll
            for (int r = 0; r < 8; ++r) {
                long row = gbase + m * 16 + rOff + r;
                out[row * 256 + col] = acc[t][m][r] + bb;
            }
    }
}

// ---------------------------------------------------------------------------
// Host launch
// ---------------------------------------------------------------------------
extern "C" void kernel_launch(void* const* d_in, const int* in_sizes, int n_in,
                              void* d_out, int out_size, void* d_ws, size_t ws_size,
                              hipStream_t stream) {
    const float* data = (const float*)d_in[0];
    const float* W1_0 = (const float*)d_in[1];
    const float* b1_0 = (const float*)d_in[2];
    const float* W1_1 = (const float*)d_in[3];
    const float* b1_1 = (const float*)d_in[4];
    const float* W2_0 = (const float*)d_in[5];
    const float* b2_0 = (const float*)d_in[6];
    const float* W2_1 = (const float*)d_in[7];
    const float* b2_1 = (const float*)d_in[8];
    const float* W3_0 = (const float*)d_in[9];
    const float* b3_0 = (const float*)d_in[10];
    const float* W3_1 = (const float*)d_in[11];
    const float* b3_1 = (const float*)d_in[12];
    const float* W4_0 = (const float*)d_in[13];
    const float* b4_0 = (const float*)d_in[14];
    const float* W4_1 = (const float*)d_in[15];
    const float* b4_1 = (const float*)d_in[16];
    const float* Wfc  = (const float*)d_in[17];
    const float* bfc  = (const float*)d_in[18];
    float* out = (float*)d_out;

    // workspace layout
    const size_t fbytes = (size_t)NT * D * sizeof(_Float16);   // 134 MB each
    _Float16* fA = (_Float16*)d_ws;
    _Float16* fB = (_Float16*)((char*)d_ws + fbytes);
    _Float16* wp = (_Float16*)((char*)d_ws + 2 * fbytes);
    _Float16* W11p = wp;                       // 128x128 -> 16384 halves
    _Float16* W20p = W11p + 16384;             // 256x128 -> 32768
    _Float16* W21p = W20p + 32768;             // 128x128
    _Float16* W30p = W21p + 16384;             // 256x128
    _Float16* W31p = W30p + 32768;             // 128x128
    _Float16* W40p = W31p + 16384;             // 256x128
    _Float16* W41p = W40p + 32768;             // 128x128
    _Float16* Wfcp = W41p + 16384;             // 128x256 -> 32768

    // repack weights into WMMA B-fragment layout (grid = (K/32)*(Nout/16))
    pack_b_kernel<<<4 * 8, 256, 0, stream>>>(W1_1, W11p, 128);
    pack_b_kernel<<<8 * 8, 256, 0, stream>>>(W2_0, W20p, 128);
    pack_b_kernel<<<4 * 8, 256, 0, stream>>>(W2_1, W21p, 128);
    pack_b_kernel<<<8 * 8, 256, 0, stream>>>(W3_0, W30p, 128);
    pack_b_kernel<<<4 * 8, 256, 0, stream>>>(W3_1, W31p, 128);
    pack_b_kernel<<<8 * 8, 256, 0, stream>>>(W4_0, W40p, 128);
    pack_b_kernel<<<4 * 8, 256, 0, stream>>>(W4_1, W41p, 128);
    pack_b_kernel<<<4 * 16, 256, 0, stream>>>(Wfc, Wfcp, 256);

    const int nblk = (int)(NT / MT);   // 8192

    layer1_kernel<<<nblk, 256, 0, stream>>>(data, W1_0, b1_0, W11p, b1_1, fA);
    res_block_kernel<<<nblk, 256, 0, stream>>>(data, fA, fB, W20p, b2_0, W21p, b2_1);
    res_block_kernel<<<nblk, 256, 0, stream>>>(data, fB, fA, W30p, b3_0, W31p, b3_1);
    res_block_kernel<<<nblk, 256, 0, stream>>>(data, fA, fB, W40p, b4_0, W41p, b4_1);
    fc_kernel<<<nblk, 256, 0, stream>>>(fB, Wfcp, bfc, out);
}
